// MultiHeadedAttention_5196910428339
// MI455X (gfx1250) — matrix operand order repaired and checked by reading, not compiled, not run on hardware
//
#include <hip/hip_runtime.h>
#include <hip/hip_bf16.h>

// MHA forward for MI455X (gfx1250): wave32, V_WMMA_F32_16X16X32_F16,
// async global->LDS staging (ASYNCcnt) with double buffering in attention.

typedef __attribute__((ext_vector_type(16))) _Float16 v16h;
typedef __attribute__((ext_vector_type(8)))  _Float16 v8h;
typedef __attribute__((ext_vector_type(8)))  float    v8f;
typedef __attribute__((ext_vector_type(4)))  float    v4f;
typedef int i32x4 __attribute__((vector_size(16)));

#define DEVI __device__ __forceinline__

constexpr int Bc = 2, Sc = 2048, Dc = 1024, Hc = 16, DKc = 64;
constexpr int NROWS = Bc * Sc;   // 4096

#if defined(__has_builtin)
#if __has_builtin(__builtin_amdgcn_global_load_async_to_lds_b128)
#define HAVE_ASYNC_LDS 1
#endif
#endif
#ifndef HAVE_ASYNC_LDS
#define HAVE_ASYNC_LDS 0
#endif

DEVI void async_copy16(const void* g, void* l) {
#if HAVE_ASYNC_LDS
  __builtin_amdgcn_global_load_async_to_lds_b128(
      (__attribute__((address_space(1))) i32x4*)g,
      (__attribute__((address_space(3))) i32x4*)l, 0, 0);
#else
  *(v8h*)l = *(const v8h*)g;   // sync fallback
#endif
}

DEVI void wait_async_lds() {
#if HAVE_ASYNC_LDS
#if __has_builtin(__builtin_amdgcn_s_wait_asynccnt)
  __builtin_amdgcn_s_wait_asynccnt(0);
#else
  asm volatile("s_wait_asynccnt 0" ::: "memory");
#endif
#endif
}

DEVI v8f wmma_f16(v16h a, v16h b, v8f c) {
  // 8 args: (neg_a, A, neg_b, B, c_mod, C, reuse_a, reuse_b)
  return __builtin_amdgcn_wmma_f32_16x16x32_f16(false, a, false, b, (short)0, c,
                                                false, false);
}

DEVI v16h cat8(v8h lo, v8h hi) {
  return __builtin_shufflevector(lo, hi, 0, 1, 2, 3, 4, 5, 6, 7,
                                         8, 9, 10, 11, 12, 13, 14, 15);
}

enum { OUT_F32 = 0, OUT_HEADS = 1, OUT_HEADS_T = 2 };

// Y[n,e] = (sum_k A[n,k] * W[e,k] + bias[e]) * scale;  N=4096, E=K=1024.
// Register-pipelined: next K-tile's global loads overlap the WMMA block.
template <bool A_IS_F32, int MODE>
__global__ __launch_bounds__(256) void gemm_wmma_kernel(
    const void* __restrict__ Av, const float* __restrict__ W,
    const float* __restrict__ bias, float scale, float* __restrict__ outF,
    _Float16* __restrict__ outH) {
  __shared__ alignas(16) _Float16 As[128][32];
  __shared__ alignas(16) _Float16 Bs[128][32];

  const int tid  = threadIdx.x;
  const int lane = tid & 31;
  const int wave = tid >> 5;
  const int wm   = wave >> 2;   // 0..1 : 64-row slab
  const int wn   = wave & 3;    // 0..3 : 32-col slab
  const int half = lane >> 4;   // 0/1
  const int l15  = lane & 15;

  const int row0 = blockIdx.y * 128;
  const int col0 = blockIdx.x * 128;

  // Staging slots: slot it covers row rS + 64*it, cols cS..cS+7.
  const int rS = tid >> 2;            // 0..63
  const int cS = (tid & 3) * 8;       // 0,8,16,24

  const float*    Af = (const float*)Av;
  const _Float16* Ah = (const _Float16*)Av;

  v8f acc[4][2];
#pragma unroll
  for (int i = 0; i < 4; ++i)
#pragma unroll
    for (int j = 0; j < 2; ++j) acc[i][j] = {};

  // Pipeline registers: raw tile data (fp32 or f16), converted at store time.
  v4f aR32[2][2], wR[2][2];
  v8h aR16[2];

  auto load_tile = [&](int k0) {
#pragma unroll
    for (int it = 0; it < 2; ++it) {
      const int r = rS + 64 * it;
      if constexpr (A_IS_F32) {
        const float* p = Af + (size_t)(row0 + r) * Dc + k0 + cS;
        aR32[it][0] = *(const v4f*)p;
        aR32[it][1] = *(const v4f*)(p + 4);
      } else {
        aR16[it] = *(const v8h*)(Ah + (size_t)(row0 + r) * Dc + k0 + cS);
      }
      const float* pw = W + (size_t)(col0 + r) * Dc + k0 + cS;
      wR[it][0] = *(const v4f*)pw;
      wR[it][1] = *(const v4f*)(pw + 4);
    }
  };

  load_tile(0);

  for (int k0 = 0; k0 < Dc; k0 += 32) {
    __syncthreads();   // previous iteration's LDS readers are done
#pragma unroll
    for (int it = 0; it < 2; ++it) {
      const int r = rS + 64 * it;
      v8h hv, wv;
      if constexpr (A_IS_F32) {
#pragma unroll
        for (int q = 0; q < 4; ++q) {
          hv[q] = (_Float16)aR32[it][0][q]; hv[q + 4] = (_Float16)aR32[it][1][q];
        }
      } else {
        hv = aR16[it];
      }
#pragma unroll
      for (int q = 0; q < 4; ++q) {
        wv[q] = (_Float16)wR[it][0][q]; wv[q + 4] = (_Float16)wR[it][1][q];
      }
      *(v8h*)(&As[r][cS]) = hv;
      *(v8h*)(&Bs[r][cS]) = wv;
    }
    __syncthreads();   // tile visible

    if (k0 + 32 < Dc) load_tile(k0 + 32);   // overlap with WMMA below

    // A fragments: lane l15 = row; lanes<16 hold K 0-7 & 16-23, lanes>=16 hold 8-15 & 24-31.
    v16h afrag[4];
#pragma unroll
    for (int i = 0; i < 4; ++i) {
      const int r  = wm * 64 + i * 16 + l15;
      const int kA = half * 8;
      v8h lo = *(const v8h*)(&As[r][kA]);
      v8h hi = *(const v8h*)(&As[r][kA + 16]);
      afrag[i] = cat8(lo, hi);
    }
    // B fragments: lane l15 = feature col; lanes<16 hold K 0-7 & 16-23, lanes>=16 hold 8-15 & 24-31.
    v16h bfrag[2];
#pragma unroll
    for (int j = 0; j < 2; ++j) {
      const int e  = wn * 32 + j * 16 + l15;
      const int kB = half * 8;
      v8h lo = *(const v8h*)(&Bs[e][kB]);
      v8h hi = *(const v8h*)(&Bs[e][kB + 16]);
      bfrag[j] = cat8(lo, hi);
    }
#pragma unroll
    for (int i = 0; i < 4; ++i)
#pragma unroll
      for (int j = 0; j < 2; ++j) acc[i][j] = wmma_f16(afrag[i], bfrag[j], acc[i][j]);
  }

  // Epilogue: C layout: VGPR v -> M = v + 8*half ; N = l15.
#pragma unroll
  for (int i = 0; i < 4; ++i) {
#pragma unroll
    for (int j = 0; j < 2; ++j) {
      const int e    = col0 + wn * 32 + j * 16 + l15;
      const float bv = bias[e];
#pragma unroll
      for (int v = 0; v < 8; ++v) {
        const int   r   = row0 + wm * 64 + i * 16 + v + half * 8;
        const float val = (acc[i][j][v] + bv) * scale;
        if constexpr (MODE == OUT_F32) {
          outF[(size_t)r * Dc + e] = val;
        } else {
          const int bb = r >> 11;       // r / S
          const int s  = r & (Sc - 1);  // r % S
          const int h  = e >> 6;        // e / DK
          const int dk = e & 63;        // e % DK
          size_t off;
          if constexpr (MODE == OUT_HEADS)
            off = (((size_t)(bb * Hc + h)) * Sc + s) * DKc + dk;   // [B,H,S,DK]
          else
            off = (((size_t)(bb * Hc + h)) * DKc + dk) * Sc + s;   // [B,H,DK,S]
          outH[off] = (_Float16)val;
        }
      }
    }
  }
}

// Flash attention: block = one (b,h) x 128 query rows; 8 waves x 16 rows.
// 64-key chunks, double-buffered in LDS via async global->LDS loads.
// Q is pre-scaled by 1/sqrt(DK) in its projection epilogue.
__global__ __launch_bounds__(256) void attn_flash_kernel(
    const _Float16* __restrict__ Q, const _Float16* __restrict__ K,
    const _Float16* __restrict__ VT, const int* __restrict__ mask,
    _Float16* __restrict__ X) {
  __shared__ alignas(16) _Float16 Ks[2][64][64];    // 64 keys x 64 d
  __shared__ alignas(16) _Float16 Vs[2][64][64];    // 64 d x 64 keys (V^T)
  __shared__ alignas(16) _Float16 Pb[8][16][64];    // per-wave P staging
  __shared__ int Ms[2][64];

  const int tid  = threadIdx.x;
  const int lane = tid & 31;
  const int wave = tid >> 5;
  const int half = lane >> 4;
  const int l15  = lane & 15;

  const int QB = Sc / 128;                     // 16 q-blocks per (b,h)
  const int qb = blockIdx.x % QB;
  const int h  = (blockIdx.x / QB) % Hc;
  const int b  = blockIdx.x / (QB * Hc);

  const size_t headSD = ((size_t)(b * Hc + h)) * Sc * DKc;  // Q/K head base
  const size_t headDS = ((size_t)(b * Hc + h)) * DKc * Sc;  // V^T head base

  const int qrow = qb * 128 + wave * 16 + l15;

  // Staging: chunk is 64x64 halves = 4096 halves -> 16 per thread (2 x b128).
  auto stage = [&](int key0, int buf) {
#pragma unroll
    for (int it = 0; it < 2; ++it) {
      const int slot = tid + it * 256;      // 0..511
      const int r = slot >> 3;              // 0..63
      const int c = (slot & 7) * 8;         // 0..56
      async_copy16(K + headSD + (size_t)(key0 + r) * DKc + c, &Ks[buf][r][c]);
      async_copy16(VT + headDS + (size_t)r * Sc + key0 + c, &Vs[buf][r][c]);
    }
    if (tid < 64) Ms[buf][tid] = mask[b * Sc + key0 + tid];
  };

  // Preload Q A-fragments (2 x 16x32 over DK=64); rows contiguous in memory.
  const _Float16* qp = Q + headSD + (size_t)qrow * DKc;
  v16h qfrag[2];
#pragma unroll
  for (int c = 0; c < 2; ++c) {
    const int off = c * 32 + half * 8;
    v8h lo = *(const v8h*)(qp + off);
    v8h hi = *(const v8h*)(qp + off + 16);
    qfrag[c] = cat8(lo, hi);
  }

  float rm[8], rsP[8];    // running row max; per-lane partial row sum
  v8f oacc[4];
#pragma unroll
  for (int v = 0; v < 8; ++v) { rm[v] = -1.0e30f; rsP[v] = 0.0f; }
#pragma unroll
  for (int t = 0; t < 4; ++t) oacc[t] = {};

  stage(0, 0);
  wait_async_lds();
  __syncthreads();

  constexpr int NCH = Sc / 64;     // 32 key chunks
  for (int ci = 0; ci < NCH; ++ci) {
    const int cur = ci & 1;
    if (ci + 1 < NCH) stage(64 * (ci + 1), cur ^ 1);   // async prefetch, no wait

    // Scores: 16 q-rows x 64 keys = four 16x16 tiles; contract DK in 2 steps.
    v8f s[4];
#pragma unroll
    for (int j = 0; j < 4; ++j) s[j] = {};
#pragma unroll
    for (int j = 0; j < 4; ++j) {
      const int krow = j * 16 + l15;          // B lane = key column
#pragma unroll
      for (int c = 0; c < 2; ++c) {
        const int off = c * 32 + half * 8;
        v8h lo = *(const v8h*)(&Ks[cur][krow][off]);
        v8h hi = *(const v8h*)(&Ks[cur][krow][off + 16]);
        s[j] = wmma_f16(qfrag[c], cat8(lo, hi), s[j]);
      }
    }
    // Mask (scale already folded into Q).
#pragma unroll
    for (int j = 0; j < 4; ++j) {
      const int mj = Ms[cur][j * 16 + l15];
#pragma unroll
      for (int v = 0; v < 8; ++v)
        s[j][v] = mj ? s[j][v] : -1.0e9f;
    }
    // Chunk row max: row r = v + 8*half lives in one 16-lane half, so
    // xor-reductions with masks 1,2,4,8 stay inside the correct lanes.
    float cm[8];
#pragma unroll
    for (int v = 0; v < 8; ++v)
      cm[v] = fmaxf(fmaxf(s[0][v], s[1][v]), fmaxf(s[2][v], s[3][v]));
#pragma unroll
    for (int m = 1; m < 16; m <<= 1)
#pragma unroll
      for (int v = 0; v < 8; ++v) cm[v] = fmaxf(cm[v], __shfl_xor(cm[v], m, 32));

    // Online update: p in place; per-lane partial sums (sf replicated per half).
#pragma unroll
    for (int v = 0; v < 8; ++v) {
      const float nm = fmaxf(rm[v], cm[v]);
      const float sf = __expf(rm[v] - nm);
      rm[v] = nm;
#pragma unroll
      for (int j = 0; j < 4; ++j) s[j][v] = __expf(s[j][v] - nm);
      rsP[v] = rsP[v] * sf + ((s[0][v] + s[1][v]) + (s[2][v] + s[3][v]));
#pragma unroll
      for (int t = 0; t < 4; ++t) oacc[t][v] *= sf;
    }

    // C-layout P -> LDS -> A-layout fragments (per-wave buffer).
#pragma unroll
    for (int j = 0; j < 4; ++j)
#pragma unroll
      for (int v = 0; v < 8; ++v)
        Pb[wave][v + half * 8][j * 16 + l15] = (_Float16)s[j][v];
    asm volatile("s_wait_dscnt 0" ::: "memory");  // same-wave LDS RAW across lanes

    v16h pfrag[2];
#pragma unroll
    for (int c = 0; c < 2; ++c) {
      const int off = c * 32 + half * 8;
      v8h lo = *(const v8h*)(&Pb[wave][l15][off]);
      v8h hi = *(const v8h*)(&Pb[wave][l15][off + 16]);
      pfrag[c] = cat8(lo, hi);
    }

    // O += P(16x64) * V(64x16) per 16-wide d tile; V^T rows = contiguous B frags.
#pragma unroll
    for (int t = 0; t < 4; ++t) {
      const int d = t * 16 + l15;
#pragma unroll
      for (int c = 0; c < 2; ++c) {
        const int koff = c * 32 + half * 8;
        v8h vlo = *(const v8h*)(&Vs[cur][d][koff]);
        v8h vhi = *(const v8h*)(&Vs[cur][d][koff + 16]);
        oacc[t] = wmma_f16(pfrag[c], cat8(vlo, vhi), oacc[t]);
      }
    }

    wait_async_lds();   // next chunk's async loads have landed
    __syncthreads();
  }

  // Final row-sum reduction (once, instead of per chunk).
#pragma unroll
  for (int m = 1; m < 16; m <<= 1)
#pragma unroll
    for (int v = 0; v < 8; ++v) rsP[v] += __shfl_xor(rsP[v], m, 32);

  // Epilogue: X[b, q, h*DK + d] = O / rowsum.
#pragma unroll
  for (int v = 0; v < 8; ++v) {
    const int   q   = qb * 128 + wave * 16 + v + half * 8;
    const float inv = 1.0f / fmaxf(rsP[v], 1.0e-30f);
#pragma unroll
    for (int t = 0; t < 4; ++t) {
      const int dcol = t * 16 + l15;
      X[((size_t)b * Sc + q) * Dc + h * DKc + dcol] = (_Float16)(oacc[t][v] * inv);
    }
  }
}

extern "C" void kernel_launch(void* const* d_in, const int* in_sizes, int n_in,
                              void* d_out, int out_size, void* d_ws,
                              size_t ws_size, hipStream_t stream) {
  const float* query = (const float*)d_in[0];
  const float* key   = (const float*)d_in[1];
  const float* value = (const float*)d_in[2];
  const int*   mask  = (const int*)d_in[3];
  const float* Wq    = (const float*)d_in[4];
  const float* bq    = (const float*)d_in[5];
  const float* Wk    = (const float*)d_in[6];
  const float* bk    = (const float*)d_in[7];
  const float* Wv    = (const float*)d_in[8];
  const float* bv    = (const float*)d_in[9];
  const float* Wo    = (const float*)d_in[10];
  const float* bo    = (const float*)d_in[11];

  // Workspace: 4 x 8 MB f16 buffers = 32 MB.
  _Float16* Qb = (_Float16*)d_ws;
  _Float16* Kb = Qb + (size_t)NROWS * Dc;
  _Float16* Vb = Kb + (size_t)NROWS * Dc;
  _Float16* Xb = Vb + (size_t)NROWS * Dc;

  dim3 blk(256);
  dim3 ggrid(Dc / 128, NROWS / 128);   // 8 x 32 blocks

  // Q projection pre-scaled by 1/sqrt(DK) = 0.125 (bias included, matching ref).
  gemm_wmma_kernel<true, OUT_HEADS><<<ggrid, blk, 0, stream>>>(
      query, Wq, bq, 0.125f, nullptr, Qb);
  gemm_wmma_kernel<true, OUT_HEADS><<<ggrid, blk, 0, stream>>>(
      key, Wk, bk, 1.0f, nullptr, Kb);
  gemm_wmma_kernel<true, OUT_HEADS_T><<<ggrid, blk, 0, stream>>>(
      value, Wv, bv, 1.0f, nullptr, Vb);

  attn_flash_kernel<<<dim3(Bc * Hc * (Sc / 128)), blk, 0, stream>>>(
      Qb, Kb, Vb, mask, Xb);

  gemm_wmma_kernel<false, OUT_F32><<<ggrid, blk, 0, stream>>>(
      Xb, Wo, bo, 1.0f, (float*)d_out, nullptr);
}
